// SparsTriangularUpdate_32409823215845
// MI455X (gfx1250) — matrix operand order repaired and checked by reading, not compiled, not run on hardware
//
#include <hip/hip_runtime.h>
#include <hip/hip_bf16.h>

typedef __bf16 bf16_t;
typedef __attribute__((ext_vector_type(16))) __bf16 v16bf;
typedef __attribute__((ext_vector_type(8)))  __bf16 v8bf;
typedef __attribute__((ext_vector_type(4)))  __bf16 v4bf;
typedef __attribute__((ext_vector_type(8)))  float  v8f;

#define LDIM 512
#define CDIM 128
#define NPOS (LDIM * LDIM)          /* 262144 positions    */
#define NELEM ((size_t)NPOS * CDIM) /* 33,554,432 elements */
#define EPSV 1e-5f

// ---------------------------------------------------------------------------
// Stage 0: one-time f32 -> bf16 weight conversion (kills v_cvt in hot loops)
// ---------------------------------------------------------------------------
__global__ void convert_weights(const float* __restrict__ W, bf16_t* __restrict__ out) {
    int i = (blockIdx.x * 256 + threadIdx.x) * 4;
    float4 v = *(const float4*)(W + i);
    v4bf o;
    o[0] = (bf16_t)v.x; o[1] = (bf16_t)v.y; o[2] = (bf16_t)v.z; o[3] = (bf16_t)v.w;
    *(v4bf*)(out + i) = o;
}

// ---------------------------------------------------------------------------
// Stage 1: global mean/var of z (two-pass deterministic tree reduction)
// ---------------------------------------------------------------------------
__global__ void zstats_partial(const float* __restrict__ z, float* __restrict__ part) {
    __shared__ float rs[256], rq[256];
    float s = 0.f, q = 0.f;
    for (size_t i = (size_t)blockIdx.x * 256 + threadIdx.x; i < NELEM;
         i += (size_t)gridDim.x * 256) {
        float v = z[i];
        s += v; q += v * v;
    }
    rs[threadIdx.x] = s; rq[threadIdx.x] = q;
    __syncthreads();
    for (int o = 128; o > 0; o >>= 1) {
        if (threadIdx.x < (unsigned)o) {
            rs[threadIdx.x] += rs[threadIdx.x + o];
            rq[threadIdx.x] += rq[threadIdx.x + o];
        }
        __syncthreads();
    }
    if (threadIdx.x == 0) { part[2 * blockIdx.x] = rs[0]; part[2 * blockIdx.x + 1] = rq[0]; }
}

// Reduce partials -> affine (scale, shift): zn = s*z + t
__global__ void stats_final(const float* __restrict__ part, int nblk,
                            const float* __restrict__ alpha, const float* __restrict__ beta,
                            float* __restrict__ stats /* [s, t] */) {
    __shared__ float rs[256], rq[256];
    float s = 0.f, q = 0.f;
    for (int i = threadIdx.x; i < nblk; i += 256) { s += part[2 * i]; q += part[2 * i + 1]; }
    rs[threadIdx.x] = s; rq[threadIdx.x] = q;
    __syncthreads();
    for (int o = 128; o > 0; o >>= 1) {
        if (threadIdx.x < (unsigned)o) {
            rs[threadIdx.x] += rs[threadIdx.x + o];
            rq[threadIdx.x] += rq[threadIdx.x + o];
        }
        __syncthreads();
    }
    if (threadIdx.x == 0) {
        float invN = 1.f / (float)NELEM;
        float m = rs[0] * invN;
        float v = rq[0] * invN - m * m;
        float r = rsqrtf(v + EPSV);
        float sc = alpha[0] * r;
        stats[0] = sc;
        stats[1] = beta[0] - m * sc;
    }
}

// ---------------------------------------------------------------------------
// Stage 2: zn = bf16(s*z + t)   (channel-last, same layout as z)
// ---------------------------------------------------------------------------
__global__ void normalize_z(const float* __restrict__ z, const float* __restrict__ stats,
                            bf16_t* __restrict__ zn) {
    float s = stats[0], t = stats[1];
    size_t i = ((size_t)blockIdx.x * 256 + threadIdx.x) * 4;
    float4 v = *(const float4*)(z + i);
    v4bf o;
    o[0] = (bf16_t)(v.x * s + t);
    o[1] = (bf16_t)(v.y * s + t);
    o[2] = (bf16_t)(v.z * s + t);
    o[3] = (bf16_t)(v.w * s + t);
    *(v4bf*)(zn + i) = o;
}

// ---------------------------------------------------------------------------
// Stage 3: a = sigmoid(zn@Wga^T + bga) * (zn@Wla^T + bla), same for b.
// Outputs stored CHANNEL-PLANAR: out[c][pos], so each lane's 8 M-values are
// one contiguous 16B store, and the einsum can load tiles without LDS.
// ---------------------------------------------------------------------------
__global__ void __launch_bounds__(256)
proj_ab(const bf16_t* __restrict__ zn,
        const bf16_t* __restrict__ Wla, const float* __restrict__ bla,
        const bf16_t* __restrict__ Wga, const float* __restrict__ bga,
        const bf16_t* __restrict__ Wlb, const float* __restrict__ blb,
        const bf16_t* __restrict__ Wgb, const float* __restrict__ bgb,
        bf16_t* __restrict__ aout, bf16_t* __restrict__ bout) {
    int lane = threadIdx.x & 31;
    int wave = threadIdx.x >> 5;
    int unit = blockIdx.x * 8 + wave;     // 262144 wave-units total
    int ab = unit & 1;
    int octile = (unit >> 1) & 7;
    int ptile = unit >> 4;
    int p0 = ptile * 16;
    int oc0 = octile * 16;

    const bf16_t* Wg = ab ? Wgb : Wga;
    const bf16_t* Wl = ab ? Wlb : Wla;
    const float*  bg = ab ? bgb : bga;
    const float*  bl = ab ? blb : bla;
    bf16_t* outp = ab ? bout : aout;

    int m = lane & 15;
    int kofs = (lane < 16) ? 0 : 16;      // A 16x32 bf16: contiguous k = 16*h + s, as B
    int n = lane & 15;
    int kq = (lane < 16) ? 0 : 16;        // B 32x16 bf16 layout

    v8f accg = {}; v8f accl = {};
#pragma unroll
    for (int kb = 0; kb < CDIM; kb += 32) {
        const bf16_t* xr = zn + (size_t)(p0 + m) * CDIM + kb + kofs;
        v8bf a0 = *(const v8bf*)(xr);
        v8bf a1 = *(const v8bf*)(xr + 8);
        v16bf Av;
#pragma unroll
        for (int e = 0; e < 8; e++) { Av[e] = a0[e]; Av[8 + e] = a1[e]; }

        const bf16_t* wg = Wg + (size_t)(oc0 + n) * CDIM + kb + kq;
        const bf16_t* wl = Wl + (size_t)(oc0 + n) * CDIM + kb + kq;
        v8bf g0 = *(const v8bf*)(wg), g1 = *(const v8bf*)(wg + 8);
        v8bf l0 = *(const v8bf*)(wl), l1 = *(const v8bf*)(wl + 8);
        v16bf Bg, Bl;
#pragma unroll
        for (int e = 0; e < 8; e++) {
            Bg[e] = g0[e]; Bg[8 + e] = g1[e];
            Bl[e] = l0[e]; Bl[8 + e] = l1[e];
        }
        accg = __builtin_amdgcn_wmma_f32_16x16x32_bf16(false, Av, false, Bg,
                                                       (short)0, accg, false, false);
        accl = __builtin_amdgcn_wmma_f32_16x16x32_bf16(false, Av, false, Bl,
                                                       (short)0, accl, false, false);
    }
    float bgv = bg[oc0 + n];
    float blv = bl[oc0 + n];
    v8bf pk;
#pragma unroll
    for (int r = 0; r < 8; r++) {
        float g = accg[r] + bgv;
        float l = accl[r] + blv;
        pk[r] = (bf16_t)(l / (1.f + __expf(-g)));
    }
    // planar store: plane (oc0+n), positions p0 .. p0+7 (lanes<16) / p0+8..15
    bf16_t* dst = outp + (size_t)(oc0 + n) * NPOS + p0 + ((lane < 16) ? 0 : 8);
    *(v8bf*)dst = pk;
}

// ---------------------------------------------------------------------------
// Stage 4: k[i,j,c] = sum_t a[i,t,c]*b[j,t,c].  a/b are channel-planar, so
// each wave owns (i-tile, j-tile, c) and streams tiles straight from
// global/L2 with b128 loads -- no LDS staging.  f32 stats fused.
// ---------------------------------------------------------------------------
__global__ void __launch_bounds__(256)
tri_einsum(const bf16_t* __restrict__ A, const bf16_t* __restrict__ B,
           bf16_t* __restrict__ K, float* __restrict__ partK) {
    __shared__ float rs[256], rq[256];

    int lane = threadIdx.x & 31;
    int wave = threadIdx.x >> 5;
    int unit = blockIdx.x * 8 + wave;     // 128*32*32 = 131072 units
    int c  = unit & 127;
    int jt = (unit >> 7) & 31;
    int it = unit >> 12;
    int i0 = it * 16, j0 = jt * 16;

    int m = lane & 15;
    int kofs = (lane < 16) ? 0 : 16;
    int n = lane & 15;
    int kq = (lane < 16) ? 0 : 16;

    const bf16_t* Ap = A + (size_t)c * NPOS + (size_t)(i0 + m) * LDIM;
    const bf16_t* Bp = B + (size_t)c * NPOS + (size_t)(j0 + n) * LDIM;

    v8f acc = {};
    for (int t0 = 0; t0 < LDIM; t0 += 32) {
        v8bf a0 = *(const v8bf*)(Ap + t0 + kofs);
        v8bf a1 = *(const v8bf*)(Ap + t0 + kofs + 8);
        v8bf b0 = *(const v8bf*)(Bp + t0 + kq);
        v8bf b1 = *(const v8bf*)(Bp + t0 + kq + 8);
        v16bf Av, Bv;
#pragma unroll
        for (int e = 0; e < 8; e++) {
            Av[e] = a0[e]; Av[8 + e] = a1[e];
            Bv[e] = b0[e]; Bv[8 + e] = b1[e];
        }
        acc = __builtin_amdgcn_wmma_f32_16x16x32_bf16(false, Av, false, Bv,
                                                      (short)0, acc, false, false);
    }

    float s = 0.f, q = 0.f;
#pragma unroll
    for (int r = 0; r < 8; r++) {
        int M = (lane < 16) ? r : r + 8;
        float v = acc[r];
        s += v; q += v * v;
        K[((size_t)(i0 + M) * LDIM + j0 + n) * CDIM + c] = (bf16_t)v;  // channel-last
    }
    rs[threadIdx.x] = s; rq[threadIdx.x] = q;
    __syncthreads();
    for (int o = 128; o > 0; o >>= 1) {
        if (threadIdx.x < (unsigned)o) {
            rs[threadIdx.x] += rs[threadIdx.x + o];
            rq[threadIdx.x] += rq[threadIdx.x + o];
        }
        __syncthreads();
    }
    if (threadIdx.x == 0) { partK[2 * blockIdx.x] = rs[0]; partK[2 * blockIdx.x + 1] = rq[0]; }
}

// ---------------------------------------------------------------------------
// Stage 5: reduce k stats; fold norm into (s2, per-channel offset) since
// Lin(s2*k + t2, W) = s2*(k@W^T) + t2*rowsum(W) + bias.
// ---------------------------------------------------------------------------
__global__ void kstats_final(const float* __restrict__ partK,
                             const float* __restrict__ alpha_o, const float* __restrict__ beta_o,
                             const float* __restrict__ Wlo, const float* __restrict__ blo,
                             float* __restrict__ stats, float* __restrict__ off) {
    __shared__ float rs[256], rq[256];
    __shared__ float tshift;
    float s = 0.f, q = 0.f;
    for (int i = threadIdx.x; i < 16384; i += 256) { s += partK[2 * i]; q += partK[2 * i + 1]; }
    rs[threadIdx.x] = s; rq[threadIdx.x] = q;
    __syncthreads();
    for (int o = 128; o > 0; o >>= 1) {
        if (threadIdx.x < (unsigned)o) {
            rs[threadIdx.x] += rs[threadIdx.x + o];
            rq[threadIdx.x] += rq[threadIdx.x + o];
        }
        __syncthreads();
    }
    if (threadIdx.x == 0) {
        float invN = 1.f / (float)NELEM;
        float m = rs[0] * invN;
        float v = rq[0] * invN - m * m;
        float r = rsqrtf(v + EPSV);
        float sc = alpha_o[0] * r;
        float tt = beta_o[0] - m * sc;
        stats[0] = sc; stats[1] = tt;
        tshift = tt;
    }
    __syncthreads();
    if (threadIdx.x < 128) {
        float rsum = 0.f;
        for (int kk = 0; kk < 128; kk++) rsum += Wlo[threadIdx.x * 128 + kk];
        off[threadIdx.x] = tshift * rsum + blo[threadIdx.x];
    }
}

// ---------------------------------------------------------------------------
// Stage 6: out = sigmoid(zn@Wgo^T + bgo) * (s2*(k@Wlo^T) + off)
// ---------------------------------------------------------------------------
__global__ void __launch_bounds__(256)
final_out(const bf16_t* __restrict__ zn, const bf16_t* __restrict__ Kb,
          const bf16_t* __restrict__ Wgo, const float* __restrict__ bgo,
          const bf16_t* __restrict__ Wlo,
          const float* __restrict__ stats2, const float* __restrict__ off,
          float* __restrict__ out) {
    int lane = threadIdx.x & 31;
    int wave = threadIdx.x >> 5;
    int unit = blockIdx.x * 8 + wave;     // 131072 units
    int octile = unit & 7;
    int ptile = unit >> 3;
    int p0 = ptile * 16;
    int oc0 = octile * 16;

    int m = lane & 15;
    int kofs = (lane < 16) ? 0 : 16;
    int n = lane & 15;
    int kq = (lane < 16) ? 0 : 16;

    v8f accg = {}; v8f accl = {};
#pragma unroll
    for (int kb = 0; kb < CDIM; kb += 32) {
        const bf16_t* xz = zn + (size_t)(p0 + m) * CDIM + kb + kofs;
        const bf16_t* xk = Kb + (size_t)(p0 + m) * CDIM + kb + kofs;
        v8bf z0 = *(const v8bf*)xz, z1 = *(const v8bf*)(xz + 8);
        v8bf k0 = *(const v8bf*)xk, k1 = *(const v8bf*)(xk + 8);
        v16bf Az, Ak;
#pragma unroll
        for (int e = 0; e < 8; e++) {
            Az[e] = z0[e]; Az[8 + e] = z1[e];
            Ak[e] = k0[e]; Ak[8 + e] = k1[e];
        }
        const bf16_t* wg = Wgo + (size_t)(oc0 + n) * CDIM + kb + kq;
        const bf16_t* wl = Wlo + (size_t)(oc0 + n) * CDIM + kb + kq;
        v8bf g0 = *(const v8bf*)(wg), g1 = *(const v8bf*)(wg + 8);
        v8bf l0 = *(const v8bf*)(wl), l1 = *(const v8bf*)(wl + 8);
        v16bf Bg, Bl;
#pragma unroll
        for (int e = 0; e < 8; e++) {
            Bg[e] = g0[e]; Bg[8 + e] = g1[e];
            Bl[e] = l0[e]; Bl[8 + e] = l1[e];
        }
        accg = __builtin_amdgcn_wmma_f32_16x16x32_bf16(false, Az, false, Bg,
                                                       (short)0, accg, false, false);
        accl = __builtin_amdgcn_wmma_f32_16x16x32_bf16(false, Ak, false, Bl,
                                                       (short)0, accl, false, false);
    }
    float s2 = stats2[0];
    float bgv = bgo[oc0 + n];
    float offv = off[oc0 + n];
#pragma unroll
    for (int r = 0; r < 8; r++) {
        int M = (lane < 16) ? r : r + 8;
        float g = accg[r] + bgv;
        float l = s2 * accl[r] + offv;
        out[(size_t)(p0 + M) * CDIM + oc0 + n] = l / (1.f + __expf(-g));
    }
}

// ---------------------------------------------------------------------------
extern "C" void kernel_launch(void* const* d_in, const int* in_sizes, int n_in,
                              void* d_out, int out_size, void* d_ws, size_t ws_size,
                              hipStream_t stream) {
    const float* z        = (const float*)d_in[0];
    const float* alpha_in = (const float*)d_in[1];
    const float* beta_in  = (const float*)d_in[2];
    const float* W_la = (const float*)d_in[3];  const float* b_la = (const float*)d_in[4];
    const float* W_ga = (const float*)d_in[5];  const float* b_ga = (const float*)d_in[6];
    const float* W_lb = (const float*)d_in[7];  const float* b_lb = (const float*)d_in[8];
    const float* W_gb = (const float*)d_in[9];  const float* b_gb = (const float*)d_in[10];
    const float* alpha_o = (const float*)d_in[11];
    const float* beta_o  = (const float*)d_in[12];
    const float* W_go = (const float*)d_in[13]; const float* b_go = (const float*)d_in[14];
    const float* W_lo = (const float*)d_in[15]; const float* b_lo = (const float*)d_in[16];
    float* out = (float*)d_out;

    char* ws = (char*)d_ws;
    float*  statsZ = (float*)(ws + 0);            // 2 floats
    float*  statsK = (float*)(ws + 8);            // 2 floats
    float*  offArr = (float*)(ws + 16);           // 128 floats
    float*  partZ  = (float*)(ws + 4096);         // 2048 float2
    float*  partK  = (float*)(ws + 20480);        // 16384 float2, ends 151552
    bf16_t* wla = (bf16_t*)(ws + 262144);         // 6 x 32 KiB bf16 weights
    bf16_t* wga = wla + 16384;
    bf16_t* wlb = wga + 16384;
    bf16_t* wgb = wlb + 16384;
    bf16_t* wgo = wgb + 16384;
    bf16_t* wlo = wgo + 16384;
    bf16_t* zn = (bf16_t*)(ws + (2ull  << 20));   // 64 MiB, channel-last
    bf16_t* ab = (bf16_t*)(ws + (66ull << 20));   // 64 MiB, channel-planar
    bf16_t* bb = (bf16_t*)(ws + (130ull << 20));  // 64 MiB, channel-planar
    bf16_t* kb = (bf16_t*)(ws + (194ull << 20));  // 64 MiB, channel-last

    convert_weights<<<16, 256, 0, stream>>>(W_la, wla);
    convert_weights<<<16, 256, 0, stream>>>(W_ga, wga);
    convert_weights<<<16, 256, 0, stream>>>(W_lb, wlb);
    convert_weights<<<16, 256, 0, stream>>>(W_gb, wgb);
    convert_weights<<<16, 256, 0, stream>>>(W_go, wgo);
    convert_weights<<<16, 256, 0, stream>>>(W_lo, wlo);

    zstats_partial<<<2048, 256, 0, stream>>>(z, partZ);
    stats_final<<<1, 256, 0, stream>>>(partZ, 2048, alpha_in, beta_in, statsZ);
    normalize_z<<<32768, 256, 0, stream>>>(z, statsZ, zn);
    proj_ab<<<32768, 256, 0, stream>>>(zn, wla, b_la, wga, b_ga,
                                       wlb, b_lb, wgb, b_gb, ab, bb);
    tri_einsum<<<16384, 256, 0, stream>>>(ab, bb, kb, partK);
    kstats_final<<<1, 256, 0, stream>>>(partK, alpha_o, beta_o, W_lo, b_lo, statsK, offArr);
    final_out<<<16384, 256, 0, stream>>>(zn, kb, wgo, b_go, wlo, statsK, offArr, out);
}
